// Circuit_41936060678727
// MI455X (gfx1250) — matrix operand order repaired and checked by reading, not compiled, not run on hardware
//
#include <hip/hip_runtime.h>
#include <math.h>

// ---------------------------------------------------------------------------
// 10-qubit real circuit simulator, restructured as:
//   out[b,p] = sum_j sign_p(j) * ( normalize(pad(x))[b] . M[j,:] )^2
// where M (1024x1024, real orthogonal) is the full circuit matrix built once
// from params. Only the first 784 input columns are nonzero, so the GEMM
// contraction is trimmed to K=800 (25 steps of the 32-wide f16 WMMA).
// GEMM runs on v_wmma_f32_16x16x32_f16 (f16 in, f32 acc).
// ---------------------------------------------------------------------------

typedef _Float16 f16;
typedef _Float16 v16h __attribute__((ext_vector_type(16)));
typedef _Float16 v8h  __attribute__((ext_vector_type(8)));
typedef float    v8f  __attribute__((ext_vector_type(8)));

#define NQ     10
#define DIM    1024          // 2^NQ (output/state dimension N)
#define KDIM   800           // contraction length: 784 rounded up to mult. of 32
#define BATCH  16384
#define XCOLS  784
#define NOUT   10
#define DEPTH  8
#define LDA    40            // LDS tile row stride in halves (80B, bank-conflict free)

// ---------------------------------------------------------------------------
// Kernel 1: L2-normalize each input row, cast to f16, pad 784 -> 800 zeros.
// One 256-thread block per batch row.
// ---------------------------------------------------------------------------
__global__ __launch_bounds__(256)
void norm_cast_kernel(const float* __restrict__ x, f16* __restrict__ Xh) {
    int b = blockIdx.x;
    const float* xr = x + (size_t)b * XCOLS;
    float ss = 0.f;
    for (int k = threadIdx.x; k < XCOLS; k += 256) {
        float v = xr[k];
        ss += v * v;
    }
    // wave32 reduction, then cross-wave via LDS
    #pragma unroll
    for (int off = 16; off >= 1; off >>= 1) ss += __shfl_xor(ss, off, 32);
    __shared__ float red[8];
    int lane = threadIdx.x & 31, wid = threadIdx.x >> 5;
    if (lane == 0) red[wid] = ss;
    __syncthreads();
    if (threadIdx.x == 0) {
        float t = 0.f;
        #pragma unroll
        for (int i = 0; i < 8; ++i) t += red[i];
        red[0] = rsqrtf(t);
    }
    __syncthreads();
    float rn = red[0];
    f16* xo = Xh + (size_t)b * KDIM;
    for (int k = threadIdx.x; k < KDIM; k += 256)
        xo[k] = (k < XCOLS) ? (f16)(xr[k] * rn) : (f16)0.f;
}

// ---------------------------------------------------------------------------
// Kernel 2: build the circuit matrix. Block k (k < KDIM) simulates
// circuit(e_k) in LDS and writes it as column k of Mh (Mh[j*KDIM + k], f16).
// CNOT = disjoint pair swap, RY = disjoint 2x2 rotation; qubit q is bit (9-q).
// ---------------------------------------------------------------------------
__global__ __launch_bounds__(256)
void build_m_kernel(const float* __restrict__ params, f16* __restrict__ Mh) {
    __shared__ float s[DIM];
    int k = blockIdx.x;
    int tid = threadIdx.x;
    for (int j = tid; j < DIM; j += 256) s[j] = (j == k) ? 1.f : 0.f;
    __syncthreads();

    const float PI = 3.14159265358979323846f;
    for (int l = 0; l < DEPTH; ++l) {
        // CNOT ladder: pairs (0,1),(2,3),... then (1,2),(3,4),...
        for (int start = 0; start < 2; ++start) {
            for (int i = start; i < NQ - 1; i += 2) {
                int ci = NQ - 1 - i;        // control bit position
                int ti = NQ - 2 - i;        // target bit position
                int tmask = 1 << ti;
                for (int j = tid; j < DIM; j += 256) {
                    if (((j >> ci) & 1) && !((j >> ti) & 1)) {
                        float a = s[j], b = s[j ^ tmask];
                        s[j] = b;
                        s[j ^ tmask] = a;
                    }
                }
                __syncthreads();
            }
        }
        // RY(pi*tanh(w)) on every wire; use half-angle
        for (int q = 0; q < NQ; ++q) {
            float th = 0.5f * PI * tanhf(params[l * NQ + q]);
            float c = cosf(th), sn = sinf(th);
            int tq = NQ - 1 - q, mask = 1 << tq;
            for (int j = tid; j < DIM; j += 256) {
                if (!((j >> tq) & 1)) {
                    float a = s[j], b = s[j | mask];
                    s[j]        = c * a - sn * b;
                    s[j | mask] = sn * a + c * b;
                }
            }
            __syncthreads();
        }
    }
    for (int j = tid; j < DIM; j += 256)
        Mh[(size_t)j * KDIM + k] = (f16)s[j];
}

// ---------------------------------------------------------------------------
// Kernel 3: zero the output (harness poisons it with 0xAA).
// ---------------------------------------------------------------------------
__global__ void zero_kernel(float* __restrict__ out, int n) {
    int i = blockIdx.x * blockDim.x + threadIdx.x;
    if (i < n) out[i] = 0.f;
}

// ---------------------------------------------------------------------------
// Kernel 4: fused GEMM + square + signed-Z reduction.
//   S[b,j] = sum_k Xh[b,k] * Mh[j,k]   via v_wmma_f32_16x16x32_f16
//   out[b,p] += sign_p(j) * S[b,j]^2   (sign from bit 9-p of j)
// 256 threads = 8 waves (4 along M x 2 along N); WG tile 128x128; K-step 32.
// LDS rows padded to 40 halves (80B) -> conflict-free ds_load_b128 fragments.
// ---------------------------------------------------------------------------
__global__ __launch_bounds__(256)
void qgemm_kernel(const f16* __restrict__ Xh, const f16* __restrict__ Mh,
                  float* __restrict__ out) {
    __shared__ __align__(16) f16 ldsA[128 * LDA];   // [row][k]   10 KB
    __shared__ __align__(16) f16 ldsB[128 * LDA];   // [col(j)][k] 10 KB

    int mBase = blockIdx.x * 128;
    int nBase = blockIdx.y * 128;
    int tid  = threadIdx.x;
    int lane = tid & 31, wid = tid >> 5;
    int l16 = lane & 15, lk = lane >> 4;
    int waveM = wid >> 1;   // 0..3 -> 32-row band
    int waveN = wid & 1;    // 0..1 -> 64-col band

    v8f acc[2][4] = {};     // 2 m-tiles x 4 n-tiles of 16x16 f32

    for (int k0 = 0; k0 < KDIM; k0 += 32) {
        __syncthreads();    // previous iteration's LDS readers done
        // stage A (X rows) and B (Mh rows = WMMA-B columns): 512 x 16B each
        #pragma unroll
        for (int c = tid; c < 512; c += 256) {
            int row = c >> 2;
            int col = (c & 3) * 8;   // halves
            *(uint4*)&ldsA[row * LDA + col] =
                *(const uint4*)&Xh[(size_t)(mBase + row) * KDIM + k0 + col];
            *(uint4*)&ldsB[row * LDA + col] =
                *(const uint4*)&Mh[(size_t)(nBase + row) * KDIM + k0 + col];
        }
        __syncthreads();

        // Fragments per ISA 7.12.2 16-bit layouts.
        union { v16h v; v8h h[2]; } af[2], bf[4];
        #pragma unroll
        for (int i = 0; i < 2; ++i) {
            int rowT = waveM * 32 + i * 16 + l16;   // A row = M index
            // halves 0..7 -> K = 8*lk + 0..7 ; halves 8..15 -> K = 8*lk + 16..23
            af[i].h[0] = *(const v8h*)&ldsA[rowT * LDA + lk * 8];
            af[i].h[1] = *(const v8h*)&ldsA[rowT * LDA + lk * 8 + 16];
        }
        #pragma unroll
        for (int t = 0; t < 4; ++t) {
            int colT = waveN * 64 + t * 16 + l16;   // B column = N index
            // halves 0..7 -> K = 8*lk + 0..7 ; halves 8..15 -> K = 8*lk + 16..23
            bf[t].h[0] = *(const v8h*)&ldsB[colT * LDA + lk * 8];
            bf[t].h[1] = *(const v8h*)&ldsB[colT * LDA + lk * 8 + 16];
        }
        #pragma unroll
        for (int i = 0; i < 2; ++i)
            #pragma unroll
            for (int t = 0; t < 4; ++t)
                acc[i][t] = __builtin_amdgcn_wmma_f32_16x16x32_f16(
                    false, af[i].v, false, bf[t].v,
                    (short)0, acc[i][t], false, false);
    }

    // Epilogue: square, apply Z-parity signs, reduce across the 16-lane
    // column group (lanes 0-15 share rows m..m+7; lanes 16-31 share m+8..m+15),
    // then atomically accumulate into out[b][p].
    int colBand = nBase + waveN * 64;
    #pragma unroll
    for (int i = 0; i < 2; ++i) {
        #pragma unroll
        for (int r = 0; r < 8; ++r) {
            float part[NOUT];
            #pragma unroll
            for (int p = 0; p < NOUT; ++p) part[p] = 0.f;
            #pragma unroll
            for (int t = 0; t < 4; ++t) {
                float v  = acc[i][t][r];
                float v2 = v * v;
                int j = colBand + t * 16 + l16;   // state index
                #pragma unroll
                for (int p = 0; p < NOUT; ++p)
                    part[p] += ((j >> (NQ - 1 - p)) & 1) ? -v2 : v2;
            }
            #pragma unroll
            for (int off = 8; off >= 1; off >>= 1)
                #pragma unroll
                for (int p = 0; p < NOUT; ++p)
                    part[p] += __shfl_xor(part[p], off, 32);
            if (l16 == 0) {
                int m = mBase + waveM * 32 + i * 16 + r + 8 * lk;
                #pragma unroll
                for (int p = 0; p < NOUT; ++p)
                    atomicAdd(&out[(size_t)m * NOUT + p], part[p]);
            }
        }
    }
}

// ---------------------------------------------------------------------------
extern "C" void kernel_launch(void* const* d_in, const int* in_sizes, int n_in,
                              void* d_out, int out_size, void* d_ws, size_t ws_size,
                              hipStream_t stream) {
    const float* x      = (const float*)d_in[0];   // [16384, 784]
    const float* params = (const float*)d_in[1];   // [8, 10]
    float* out = (float*)d_out;                    // [16384, 10]

    f16* Xh = (f16*)d_ws;                                              // 25.6 MB
    f16* Mh = (f16*)((char*)d_ws + (size_t)BATCH * KDIM * sizeof(f16)); // 1.6 MB

    norm_cast_kernel<<<BATCH, 256, 0, stream>>>(x, Xh);
    build_m_kernel<<<KDIM, 256, 0, stream>>>(params, Mh);
    zero_kernel<<<(BATCH * NOUT + 255) / 256, 256, 0, stream>>>(out, BATCH * NOUT);

    dim3 grid(BATCH / 128, DIM / 128);
    qgemm_kernel<<<grid, 256, 0, stream>>>(Xh, Mh, out);
}
